// Model_4698694222517
// MI455X (gfx1250) — matrix operand order repaired and checked by reading, not compiled, not run on hardware
//
#include <hip/hip_runtime.h>
#include <hip/hip_bf16.h>

#define EMB 300
#define NNODES 200000
#define NEDGES 440000
#define NGRAPHS 4096
#define NLAYERS 5
#define SELF_LOOP 4

typedef __attribute__((ext_vector_type(16))) __bf16 v16bf;
typedef __attribute__((ext_vector_type(8)))  float  v8f;

union BFVec { unsigned u[8]; v16bf v; };

__device__ __forceinline__ unsigned short f2bf(float x) {
  unsigned u = __float_as_uint(x);
  u += 0x7fffu + ((u >> 16) & 1u);          // round-to-nearest-even
  return (unsigned short)(u >> 16);
}

// ---------- fragment loaders (CDNA5 wave32 WMMA layouts) ----------
// A (16x32 bf16): lane l holds row m=l%16; element pair g: K = ktile*32 + (l>>4)*8 + (g<4 ? 2g : 2g+8)
__device__ __forceinline__ v16bf load_fragA(const unsigned short* base, int stride, int ktile) {
  int lane = threadIdx.x & 31;
  const unsigned short* p = base + (lane & 15) * stride + ktile * 32 + (lane >> 4) * 8;
  BFVec f;
#pragma unroll
  for (int g = 0; g < 8; ++g) {
    int k0 = (g < 4) ? 2 * g : 2 * g + 8;
    f.u[g] = *(const unsigned*)(p + k0);
  }
  return f.v;
}
// B from pre-packed buffer: 8 contiguous dwords per lane per tile
__device__ __forceinline__ v16bf load_fragB_packed(const unsigned* pack, int tileIdx) {
  int lane = threadIdx.x & 31;
  const unsigned* p = pack + ((size_t)tileIdx * 32 + lane) * 8;
  BFVec f;
#pragma unroll
  for (int g = 0; g < 8; ++g) f.u[g] = p[g];
  return f.v;
}
// B (32x16 bf16) straight from a row-major bf16 matrix where column n of B = row n of M
__device__ __forceinline__ v16bf load_fragB_rows(const unsigned short* M, int stride, int n0, int ktile) {
  int lane = threadIdx.x & 31;
  const unsigned short* p = M + (size_t)(n0 + (lane & 15)) * stride + ktile * 32 + ((lane < 16) ? 0 : 8);
  BFVec f;
#pragma unroll
  for (int g = 0; g < 8; ++g) f.u[g] = *(const unsigned*)(p + ((g < 4) ? 2 * g : 2 * g + 8));
  return f.v;
}

// ---------- weight packing: W[K,N] fp32 -> B-fragment order bf16 ----------
__global__ void pack_w(const float* __restrict__ W, int K, int N, int kt, unsigned* __restrict__ P) {
  int tile = blockIdx.x;                 // tile = ntile*kt + ktile
  int ktile = tile % kt, ntile = tile / kt;
  int lane = threadIdx.x & 31, g = threadIdx.x >> 5;
  int n  = ntile * 16 + (lane & 15);
  int K0 = ktile * 32 + ((g < 4) ? 2 * g : 2 * g + 8) + ((lane < 16) ? 0 : 8);
  float w0 = (K0     < K && n < N) ? W[(size_t)K0 * N + n]       : 0.f;
  float w1 = (K0 + 1 < K && n < N) ? W[(size_t)(K0 + 1) * N + n] : 0.f;
  P[((size_t)tile * 32 + lane) * 8 + g] = (unsigned)f2bf(w0) | ((unsigned)f2bf(w1) << 16);
}

// ---------- h = atom_emb1[x0] + atom_emb2[x1] ----------
__global__ void node_embed(const int* __restrict__ x, const float* __restrict__ a1,
                           const float* __restrict__ a2, float* __restrict__ h) {
  long i = (long)blockIdx.x * blockDim.x + threadIdx.x;
  if (i >= (long)NNODES * EMB) return;
  int node = i / EMB, c = i % EMB;
  h[i] = a1[x[node * 2] * EMB + c] + a2[x[node * 2 + 1] * EMB + c];
}

// ---------- agg = h + self_loop_edge_emb ----------
__global__ void agg_init(const float* __restrict__ h, const float* __restrict__ ee1l,
                         const float* __restrict__ ee2l, float* __restrict__ agg) {
  long i = (long)blockIdx.x * blockDim.x + threadIdx.x;
  if (i >= (long)NNODES * EMB) return;
  int c = i % EMB;
  agg[i] = h[i] + ee1l[SELF_LOOP * EMB + c] + ee2l[c];
}

// ---------- scatter: agg[dst] += h[src] + edge_emb  (wave per edge) ----------
__global__ void edge_scatter(const float* __restrict__ h, const int* __restrict__ src,
                             const int* __restrict__ dst, const int* __restrict__ ea,
                             const float* __restrict__ ee1l, const float* __restrict__ ee2l,
                             float* __restrict__ agg) {
  int e = blockIdx.x * (blockDim.x >> 5) + (threadIdx.x >> 5);
  if (e >= NEDGES) return;
  int lane = threadIdx.x & 31;
  int s = src[e], d = dst[e];
  const float* hs = h + (size_t)s * EMB;
  const float* e1 = ee1l + (size_t)ea[e * 2] * EMB;
  const float* e2 = ee2l + (size_t)ea[e * 2 + 1] * EMB;
  float* ag = agg + (size_t)d * EMB;
  for (int c = lane; c < EMB; c += 32) atomicAdd(&ag[c], hs[c] + e1[c] + e2[c]);
}

// ---------- fused MLP: Out = relu(A@W1+b1)@W2+b2, + optional BN stat atomics ----------
__global__ __launch_bounds__(256)
void fused_mlp(const float* __restrict__ A, int in_dim,
               const unsigned* __restrict__ P1, const float* __restrict__ b1,
               int nt1, int kt1, int mid_cols,
               const unsigned* __restrict__ P2, const float* __restrict__ b2,
               int nt2, int kt2,
               float* __restrict__ Out, int out_cols,
               float* __restrict__ bnstats) {
  __shared__ unsigned short Abf[16 * 320];
  __shared__ unsigned short Tm[16 * 608];
  int r0 = blockIdx.x * 16;
  int tid = threadIdx.x;
  for (int i = tid; i < 16 * 320; i += 256) {           // load+convert A tile (zero-pad K)
    int r = i / 320, c = i % 320;
    Abf[i] = f2bf((c < in_dim) ? A[(size_t)(r0 + r) * in_dim + c] : 0.f);
  }
  for (int i = tid; i < 16 * 608; i += 256) Tm[i] = 0;  // zero intermediate (pads K of GEMM2)
  __syncthreads();
  int wave = tid >> 5, lane = tid & 31;
  // GEMM1: T = relu(A@W1 + b1)
  for (int t = wave; t < nt1; t += 8) {
    v8f acc = {};
    for (int k = 0; k < kt1; ++k) {
      v16bf a = load_fragA(Abf, 320, k);
      v16bf b = load_fragB_packed(P1, t * kt1 + k);
      acc = __builtin_amdgcn_wmma_f32_16x16x32_bf16(false, a, false, b, (short)0, acc, false, false);
    }
    int n = t * 16 + (lane & 15);
    float bias = (n < mid_cols) ? b1[n] : 0.f;
    int mbase = (lane < 16) ? 0 : 8;
#pragma unroll
    for (int r = 0; r < 8; ++r) {
      float v = acc[r] + bias;
      Tm[(mbase + r) * 608 + n] = f2bf(v > 0.f ? v : 0.f);
    }
  }
  __syncthreads();
  // GEMM2: Out = T@W2 + b2
  for (int t = wave; t < nt2; t += 8) {
    v8f acc = {};
    for (int k = 0; k < kt2; ++k) {
      v16bf a = load_fragA(Tm, 608, k);
      v16bf b = load_fragB_packed(P2, t * kt2 + k);
      acc = __builtin_amdgcn_wmma_f32_16x16x32_bf16(false, a, false, b, (short)0, acc, false, false);
    }
    int n = t * 16 + (lane & 15);
    if (n < out_cols) {
      float bias = b2[n];
      int mbase = (lane < 16) ? 0 : 8;
      float s = 0.f, sq = 0.f;
#pragma unroll
      for (int r = 0; r < 8; ++r) {
        float v = acc[r] + bias;
        Out[(size_t)(r0 + mbase + r) * out_cols + n] = v;
        s += v; sq += v * v;
      }
      if (bnstats) { atomicAdd(&bnstats[n], s); atomicAdd(&bnstats[out_cols + n], sq); }
    }
  }
}

// ---------- BatchNorm ----------
__global__ void bn_finalize(const float* __restrict__ st, const float* __restrict__ gm,
                            const float* __restrict__ bt, float* __restrict__ ss, float invN) {
  int c = blockIdx.x * blockDim.x + threadIdx.x;
  if (c >= EMB) return;
  float mean = st[c] * invN;
  float var = st[EMB + c] * invN - mean * mean;
  float sc = gm[c] * rsqrtf(var + 1e-5f);
  ss[c] = sc; ss[EMB + c] = bt[c] - mean * sc;
}
__global__ void bn_apply(float* __restrict__ h, const float* __restrict__ ss, int do_relu) {
  long i = (long)blockIdx.x * blockDim.x + threadIdx.x;
  if (i >= (long)NNODES * EMB) return;
  int c = i % EMB;
  float v = h[i] * ss[c] + ss[EMB + c];
  h[i] = (do_relu && v < 0.f) ? 0.f : v;
}

// ---------- global mean pool ----------
__global__ void pool_scatter(const float* __restrict__ h, const int* __restrict__ bid,
                             float* __restrict__ s, float* __restrict__ cnt) {
  int node = blockIdx.x * (blockDim.x >> 5) + (threadIdx.x >> 5);
  if (node >= NNODES) return;
  int lane = threadIdx.x & 31;
  int b = bid[node];
  const float* hv = h + (size_t)node * EMB;
  float* sv = s + (size_t)b * EMB;
  for (int c = lane; c < EMB; c += 32) atomicAdd(&sv[c], hv[c]);
  if (lane == 0) atomicAdd(&cnt[b], 1.0f);
}
__global__ void pool_div(const float* __restrict__ s, const float* __restrict__ cnt,
                         float* __restrict__ g) {
  long i = (long)blockIdx.x * blockDim.x + threadIdx.x;
  if (i >= (long)NGRAPHS * EMB) return;
  float c = cnt[i / EMB];
  g[i] = s[i] / (c > 1.0f ? c : 1.0f);
}

// ---------- row-normalize -> bf16 [4096 x 304] ----------
__global__ __launch_bounds__(256)
void normalize_bf(const float* __restrict__ f, unsigned short* __restrict__ fb) {
  int row = blockIdx.x;
  __shared__ float red[256];
  float s = 0.f;
  for (int c = threadIdx.x; c < EMB; c += 256) { float v = f[(size_t)row * EMB + c]; s += v * v; }
  red[threadIdx.x] = s; __syncthreads();
  for (int o = 128; o > 0; o >>= 1) { if (threadIdx.x < o) red[threadIdx.x] += red[threadIdx.x + o]; __syncthreads(); }
  float inv = rsqrtf(red[0]);
  for (int c = threadIdx.x; c < 304; c += 256)
    fb[(size_t)row * 304 + c] = (c < EMB) ? f2bf(f[(size_t)row * EMB + c] * inv) : (unsigned short)0;
}

// ---------- logits = (f0 @ f1^T) / TEMP ----------
__global__ __launch_bounds__(256)
void logits_wmma(const unsigned short* __restrict__ f0, const unsigned short* __restrict__ f1,
                 float* __restrict__ out) {
  int wave = threadIdx.x >> 5, lane = threadIdx.x & 31;
  int row0 = blockIdx.x * 16;
  int col0 = (blockIdx.y * 8 + wave) * 16;
  v8f acc = {};
  for (int k = 0; k < 10; ++k) {
    int l15 = lane & 15;
    const unsigned short* pa = f0 + (size_t)(row0 + l15) * 304 + k * 32 + (lane >> 4) * 8;
    BFVec fa;
#pragma unroll
    for (int g = 0; g < 8; ++g) { int k0 = (g < 4) ? 2 * g : 2 * g + 8; fa.u[g] = *(const unsigned*)(pa + k0); }
    v16bf b = load_fragB_rows(f1, 304, col0, k);
    acc = __builtin_amdgcn_wmma_f32_16x16x32_bf16(false, fa.v, false, b, (short)0, acc, false, false);
  }
  int n = col0 + (lane & 15);
  int mbase = (lane < 16) ? 0 : 8;
#pragma unroll
  for (int r = 0; r < 8; ++r)
    out[(size_t)(row0 + mbase + r) * NGRAPHS + n] = acc[r] * 25.0f;   // 1/TEMP = 25
}

__global__ void write_labels(float* __restrict__ out) {
  int i = blockIdx.x * blockDim.x + threadIdx.x;
  if (i < NGRAPHS) out[(size_t)NGRAPHS * NGRAPHS + i] = (float)i;
}

extern "C" void kernel_launch(void* const* d_in, const int* in_sizes, int n_in,
                              void* d_out, int out_size, void* d_ws, size_t ws_size,
                              hipStream_t stream) {
  const int*   x[2]   = {(const int*)d_in[0], (const int*)d_in[4]};
  const int*   ei[2]  = {(const int*)d_in[1], (const int*)d_in[5]};
  const int*   ea[2]  = {(const int*)d_in[2], (const int*)d_in[6]};
  const int*   bid[2] = {(const int*)d_in[3], (const int*)d_in[7]};
  const float* atom1 = (const float*)d_in[8];
  const float* atom2 = (const float*)d_in[9];
  const float* ee1   = (const float*)d_in[10];
  const float* ee2   = (const float*)d_in[11];
  const float* W1    = (const float*)d_in[12];
  const float* bm1   = (const float*)d_in[13];
  const float* W2    = (const float*)d_in[14];
  const float* bm2   = (const float*)d_in[15];
  const float* gamma = (const float*)d_in[16];
  const float* beta  = (const float*)d_in[17];
  const float* Wp1   = (const float*)d_in[18];
  const float* bp1   = (const float*)d_in[19];
  const float* Wp2   = (const float*)d_in[20];
  const float* bp2   = (const float*)d_in[21];

  char* w = (char*)d_ws;
  auto alloc = [&](size_t bytes) { void* p = (void*)w; w += (bytes + 255) & ~(size_t)255; return p; };

  float* h    = (float*)alloc((size_t)NNODES * EMB * 4);
  float* agg  = (float*)alloc((size_t)NNODES * EMB * 4);
  float* bn   = (float*)alloc(2 * EMB * 4);          // sums, sumsq
  float* ss   = (float*)alloc(2 * EMB * 4);          // scale, shift
  float* psum = (float*)alloc((size_t)NGRAPHS * EMB * 4);
  float* pcnt = (float*)alloc((size_t)NGRAPHS * 4);
  float* g    = (float*)alloc((size_t)NGRAPHS * EMB * 4);
  float* fpre = (float*)alloc((size_t)NGRAPHS * EMB * 4);
  unsigned short* fbf[2];
  fbf[0] = (unsigned short*)alloc((size_t)NGRAPHS * 304 * 2);
  fbf[1] = (unsigned short*)alloc((size_t)NGRAPHS * 304 * 2);
  const size_t P1SZ = (size_t)38 * 10 * 32 * 8;      // dwords
  const size_t P2SZ = (size_t)19 * 19 * 32 * 8;
  const size_t PPSZ = (size_t)19 * 10 * 32 * 8;
  unsigned* P1 = (unsigned*)alloc(NLAYERS * P1SZ * 4);
  unsigned* P2 = (unsigned*)alloc(NLAYERS * P2SZ * 4);
  unsigned* Pp1 = (unsigned*)alloc(PPSZ * 4);
  unsigned* Pp2 = (unsigned*)alloc(PPSZ * 4);

  // pack all weights (L2-resident, reused by every block)
  for (int l = 0; l < NLAYERS; ++l) {
    pack_w<<<38 * 10, 256, 0, stream>>>(W1 + (size_t)l * 300 * 600, 300, 600, 10, P1 + l * P1SZ);
    pack_w<<<19 * 19, 256, 0, stream>>>(W2 + (size_t)l * 600 * 300, 600, 300, 19, P2 + l * P2SZ);
  }
  pack_w<<<19 * 10, 256, 0, stream>>>(Wp1, 300, 300, 10, Pp1);
  pack_w<<<19 * 10, 256, 0, stream>>>(Wp2, 300, 300, 10, Pp2);

  const int ELT_BLKS = (int)(((long)NNODES * EMB + 255) / 256);

  for (int enc = 0; enc < 2; ++enc) {
    node_embed<<<ELT_BLKS, 256, 0, stream>>>(x[enc], atom1, atom2, h);
    for (int l = 0; l < NLAYERS; ++l) {
      const float* e1l = ee1 + (size_t)l * 6 * EMB;
      const float* e2l = ee2 + (size_t)l * 3 * EMB;
      agg_init<<<ELT_BLKS, 256, 0, stream>>>(h, e1l, e2l, agg);
      edge_scatter<<<(NEDGES + 7) / 8, 256, 0, stream>>>(h, ei[enc], ei[enc] + NEDGES,
                                                         ea[enc], e1l, e2l, agg);
      hipMemsetAsync(bn, 0, 2 * EMB * 4, stream);
      fused_mlp<<<NNODES / 16, 256, 0, stream>>>(agg, EMB,
                                                 P1 + l * P1SZ, bm1 + (size_t)l * 600, 38, 10, 600,
                                                 P2 + l * P2SZ, bm2 + (size_t)l * 300, 19, 19,
                                                 h, EMB, bn);
      bn_finalize<<<(EMB + 255) / 256, 256, 0, stream>>>(bn, gamma + (size_t)l * EMB,
                                                         beta + (size_t)l * EMB, ss, 1.0f / NNODES);
      bn_apply<<<ELT_BLKS, 256, 0, stream>>>(h, ss, l < NLAYERS - 1 ? 1 : 0);
    }
    hipMemsetAsync(psum, 0, (size_t)NGRAPHS * EMB * 4, stream);
    hipMemsetAsync(pcnt, 0, (size_t)NGRAPHS * 4, stream);
    pool_scatter<<<NNODES / 8, 256, 0, stream>>>(h, bid[enc], psum, pcnt);
    pool_div<<<(NGRAPHS * EMB + 255) / 256, 256, 0, stream>>>(psum, pcnt, g);
    fused_mlp<<<NGRAPHS / 16, 256, 0, stream>>>(g, EMB, Pp1, bp1, 19, 10, 300,
                                                Pp2, bp2, 19, 10, fpre, EMB, nullptr);
    normalize_bf<<<NGRAPHS, 256, 0, stream>>>(fpre, fbf[enc]);
  }

  logits_wmma<<<dim3(NGRAPHS / 16, NGRAPHS / 128), 256, 0, stream>>>(fbf[0], fbf[1], (float*)d_out);
  write_labels<<<NGRAPHS / 256, 256, 0, stream>>>((float*)d_out);
}
